// RWKVBlock_83760452206822
// MI455X (gfx1250) — matrix operand order repaired and checked by reading, not compiled, not run on hardware
//
#include <hip/hip_runtime.h>
#include <hip/hip_bf16.h>

// ---------------------------------------------------------------------------
// RWKV-4 block for MI455X (gfx1250, wave32).
// GEMMs run through v_wmma_f32_16x16x32_bf16 (bf16 in, f32 accumulate),
// with a register double-buffered K-loop so global loads overlap the WMMAs.
// ---------------------------------------------------------------------------

typedef __bf16 bf16;
typedef bf16  v16bf __attribute__((ext_vector_type(16)));
typedef bf16  v8bf  __attribute__((ext_vector_type(8)));
typedef float v8f   __attribute__((ext_vector_type(8)));

#define RB 8           // batch
#define RT 1024        // sequence
#define RD 1024        // model dim
#define RF 4096        // ffn hidden
#define RN (RB * RT)   // total rows = 8192

// ---------------------------------------------------------------------------
// LayerNorm over last dim (D=1024), one block (256 thr) per row.
// Also emits xn[:, -1, :] for the state outputs.
// ---------------------------------------------------------------------------
__global__ __launch_bounds__(256) void ln_kernel(
    const float* __restrict__ x, const float* __restrict__ g,
    const float* __restrict__ b, float* __restrict__ xn,
    float* __restrict__ last_out)
{
    const int rowid = blockIdx.x;            // b*T + t
    const int t = rowid % RT;
    const int bi = rowid / RT;
    const float* xr = x + (size_t)rowid * RD;

    float s = 0.f, s2 = 0.f;
    for (int i = threadIdx.x; i < RD; i += 256) {
        float v = xr[i];
        s += v; s2 += v * v;
    }
    __shared__ float red[512];
    red[threadIdx.x] = s;
    red[256 + threadIdx.x] = s2;
    __syncthreads();
    for (int off = 128; off > 0; off >>= 1) {
        if (threadIdx.x < off) {
            red[threadIdx.x]       += red[threadIdx.x + off];
            red[256 + threadIdx.x] += red[256 + threadIdx.x + off];
        }
        __syncthreads();
    }
    const float mu   = red[0] * (1.0f / RD);
    const float var  = red[256] * (1.0f / RD) - mu * mu;
    const float rstd = rsqrtf(var + 1e-3f);

    float* outr = xn + (size_t)rowid * RD;
    for (int i = threadIdx.x; i < RD; i += 256) {
        float v = (xr[i] - mu) * rstd * g[i] + b[i];
        outr[i] = v;
        if (t == RT - 1) last_out[(size_t)bi * RD + i] = v;
    }
}

// ---------------------------------------------------------------------------
// Token-shift mix: out = bf16(xn[t]*mix + prev*(1-mix)), prev = state at t==0.
// ---------------------------------------------------------------------------
__global__ __launch_bounds__(256) void mix_kernel(
    const float* __restrict__ xn, const float* __restrict__ state,
    const float* __restrict__ mix, bf16* __restrict__ out)
{
    const size_t id = (size_t)blockIdx.x * 256 + threadIdx.x;   // over N*D
    const int d = (int)(id % RD);
    const size_t m = id / RD;
    const int t = (int)(m % RT);
    const int bi = (int)(m / RT);
    const float cur  = xn[id];
    const float prev = (t == 0) ? state[(size_t)bi * RD + d] : xn[id - RD];
    const float mx = mix[d];
    out[id] = (bf16)(cur * mx + prev * (1.0f - mx));
}

// ---------------------------------------------------------------------------
// Elementwise r*wkv -> bf16 (GEMM A panel for Wo).
// ---------------------------------------------------------------------------
__global__ __launch_bounds__(256) void mul_bf16_kernel(
    const float* __restrict__ a, const float* __restrict__ b,
    bf16* __restrict__ out)
{
    const size_t id = (size_t)blockIdx.x * 256 + threadIdx.x;
    out[id] = (bf16)(a[id] * b[id]);
}

// ---------------------------------------------------------------------------
// Convert f32 weight [K x Nn] -> bf16 transposed [Nn x K] so that WMMA B
// fragments become contiguous 16B loads.
// ---------------------------------------------------------------------------
__global__ __launch_bounds__(256) void convtrans_kernel(
    const float* __restrict__ w, bf16* __restrict__ wt, int K, int Nn)
{
    const size_t id = (size_t)blockIdx.x * 256 + threadIdx.x; // over Nn*K
    if (id >= (size_t)K * Nn) return;
    const int k = (int)(id % K);
    const int n = (int)(id / K);
    wt[id] = (bf16)w[(size_t)k * Nn + n];
}

// ---------------------------------------------------------------------------
// Fragment helpers.
// ---------------------------------------------------------------------------
__device__ __forceinline__ v16bf mk16(const v8bf lo, const v8bf hi) {
    v16bf r;
#pragma unroll
    for (int i = 0; i < 8; ++i) { r[i] = lo[i]; r[8 + i] = hi[i]; }
    return r;
}

// Load one A fragment (16x32, ISA 7.12.2 lane layout) + 4 B fragments.
#define LOAD_FRAGS(AL, AH, BL, BH, KOFF)                                     \
    {                                                                        \
        const bf16* ap_ = arow + (KOFF);                                     \
        AL = *(const v8bf*)ap_;                                              \
        AH = *(const v8bf*)(ap_ + 16);                                       \
        _Pragma("unroll")                                                    \
        for (int j_ = 0; j_ < 4; ++j_) {                                     \
            const bf16* bp_ = brow[j_] + (KOFF);                             \
            BL[j_] = *(const v8bf*)bp_;                                      \
            BH[j_] = *(const v8bf*)(bp_ + 16);                               \
        }                                                                    \
    }

#define COMPUTE_FRAGS(AL, AH, BL, BH)                                        \
    {                                                                        \
        const v16bf af_ = mk16(AL, AH);                                      \
        _Pragma("unroll")                                                    \
        for (int j_ = 0; j_ < 4; ++j_) {                                     \
            acc[j_] = __builtin_amdgcn_wmma_f32_16x16x32_bf16(               \
                false, af_, false, mk16(BL[j_], BH[j_]), (short)0,           \
                acc[j_], false, false);                                      \
        }                                                                    \
    }

// ---------------------------------------------------------------------------
// WMMA GEMM: C[M,Nn] = A[M,K] @ W[K,Nn], A bf16 row-major, Wt bf16 [Nn,K].
// Each wave owns a 16x64 strip of C (1 A-frag feeds 4 WMMAs per K-step);
// K-loop is double-buffered in registers (K must be a multiple of 64).
// Epilogues:
//   EPI 0: f32 store             EPI 1: sigmoid -> f32
//   EPI 2: + residual -> f32     EPI 3: relu^2 -> bf16
//   EPI 4: res + gate*acc -> f32 (final output)
// ---------------------------------------------------------------------------
template <int EPI>
__global__ __launch_bounds__(256) void gemm_wmma_kernel(
    const bf16* __restrict__ A, const bf16* __restrict__ Wt,
    int K, int Nn,
    float* __restrict__ out_f, bf16* __restrict__ out_h,
    const float* __restrict__ res, const float* __restrict__ gate)
{
    const int wave = threadIdx.x >> 5;
    const int lane = threadIdx.x & 31;
    const int half = lane >> 4;      // lane group 0..15 vs 16..31
    const int row  = lane & 15;

    const int m0 = (blockIdx.x * 8 + wave) * 16;  // 16-row tile
    const int n0 = blockIdx.y * 64;               // 64-col strip

    // Per-lane base pointers.
    const bf16* arow = A + (size_t)(m0 + row) * K + half * 8;
    const bf16* brow[4];
#pragma unroll
    for (int j = 0; j < 4; ++j)
        brow[j] = Wt + (size_t)(n0 + j * 16 + row) * K + half * 8;

    v8f acc[4] = {v8f{}, v8f{}, v8f{}, v8f{}};

    // Double-buffered fragments.
    v8bf a0l, a0h, a1l, a1h;
    v8bf b0l[4], b0h[4], b1l[4], b1h[4];

    LOAD_FRAGS(a0l, a0h, b0l, b0h, 0);

    int kb = 0;
    for (; kb < K - 64; kb += 64) {
        LOAD_FRAGS(a1l, a1h, b1l, b1h, kb + 32);   // prefetch next
        COMPUTE_FRAGS(a0l, a0h, b0l, b0h);         // consume current
        LOAD_FRAGS(a0l, a0h, b0l, b0h, kb + 64);
        COMPUTE_FRAGS(a1l, a1h, b1l, b1h);
    }
    // Tail: kb == K-64, buffer 0 holds kb, prefetch kb+32 then drain both.
    LOAD_FRAGS(a1l, a1h, b1l, b1h, kb + 32);
    COMPUTE_FRAGS(a0l, a0h, b0l, b0h);
    COMPUTE_FRAGS(a1l, a1h, b1l, b1h);

    // C layout: VGPR g holds (M = m0 + g + 8*half, N = n0 + j*16 + row).
#pragma unroll
    for (int j = 0; j < 4; ++j) {
#pragma unroll
        for (int g = 0; g < 8; ++g) {
            const int mm = m0 + g + 8 * half;
            const int nn = n0 + j * 16 + row;
            const size_t idx = (size_t)mm * Nn + nn;
            const float v = acc[j][g];
            if constexpr (EPI == 0) {
                out_f[idx] = v;
            } else if constexpr (EPI == 1) {
                out_f[idx] = 1.0f / (1.0f + __expf(-v));
            } else if constexpr (EPI == 2) {
                out_f[idx] = res[idx] + v;
            } else if constexpr (EPI == 3) {
                const float t = v > 0.0f ? v : 0.0f;
                out_h[idx] = (bf16)(t * t);
            } else { // EPI == 4
                out_f[idx] = res[idx] + gate[idx] * v;
            }
        }
    }
}

// ---------------------------------------------------------------------------
// Numerically stable WKV recurrence: one lane per (b, d) channel, T steps.
// ---------------------------------------------------------------------------
__global__ __launch_bounds__(256) void wkv_kernel(
    const float* __restrict__ k, const float* __restrict__ v,
    const float* __restrict__ a0, const float* __restrict__ b0,
    const float* __restrict__ p0, const float* __restrict__ decay,
    const float* __restrict__ first, float* __restrict__ wkv,
    float* __restrict__ aa_out, float* __restrict__ bb_out,
    float* __restrict__ pp_out)
{
    const int id = blockIdx.x * 256 + threadIdx.x;   // over B*D
    const int d = id % RD;
    const int bi = id / RD;
    float aa = a0[id], bb = b0[id], pp = p0[id];
    const float w = -__expf(decay[d]);
    const float u = first[d];
    size_t base = (size_t)bi * RT * RD + d;
    for (int t = 0; t < RT; ++t) {
        const float kt = k[base], vt = v[base];
        const float ww = u + kt;
        const float p  = fmaxf(pp, ww);
        const float e1 = __expf(pp - p);
        const float e2 = __expf(ww - p);
        wkv[base] = (e1 * aa + e2 * vt) / (e1 * bb + e2);
        const float ww2 = pp + w;
        const float p2  = fmaxf(ww2, kt);
        const float e1b = __expf(ww2 - p2);
        const float e2b = __expf(kt - p2);
        aa = e1b * aa + e2b * vt;
        bb = e1b * bb + e2b;
        pp = p2;
        base += RD;
    }
    aa_out[id] = aa;
    bb_out[id] = bb;
    pp_out[id] = pp;
}

// ---------------------------------------------------------------------------
// Host orchestration
// ---------------------------------------------------------------------------
extern "C" void kernel_launch(void* const* d_in, const int* in_sizes, int n_in,
                              void* d_out, int out_size, void* d_ws, size_t ws_size,
                              hipStream_t stream) {
    (void)in_sizes; (void)n_in; (void)out_size; (void)ws_size;

    const float* x        = (const float*)d_in[0];
    const float* att_x    = (const float*)d_in[1];
    const float* att_a    = (const float*)d_in[2];
    const float* att_b    = (const float*)d_in[3];
    const float* att_p    = (const float*)d_in[4];
    const float* ffn_x    = (const float*)d_in[5];
    const float* ln1_g    = (const float*)d_in[6];
    const float* ln1_b    = (const float*)d_in[7];
    const float* ln2_g    = (const float*)d_in[8];
    const float* ln2_b    = (const float*)d_in[9];
    const float* tm_mix_k = (const float*)d_in[10];
    const float* tm_mix_v = (const float*)d_in[11];
    const float* tm_mix_r = (const float*)d_in[12];
    const float* tm_decay = (const float*)d_in[13];
    const float* tm_first = (const float*)d_in[14];
    const float* Wk       = (const float*)d_in[15];
    const float* Wv       = (const float*)d_in[16];
    const float* Wr       = (const float*)d_in[17];
    const float* Wo       = (const float*)d_in[18];
    const float* cm_mix_k = (const float*)d_in[19];
    const float* cm_mix_r = (const float*)d_in[20];
    const float* Ck       = (const float*)d_in[21];
    const float* Cr       = (const float*)d_in[22];
    const float* Cv       = (const float*)d_in[23];

    const size_t ND = (size_t)RN * RD;     // 8,388,608
    const size_t NF = (size_t)RN * RF;     // 33,554,432
    const size_t DF = (size_t)RD * RF;     // 4,194,304
    const size_t BD = (size_t)RB * RD;     // 8,192

    // Output layout (f32): out | xn_last | aa | bb | pp | xn2_last
    float* out      = (float*)d_out;
    float* xn_last  = out + ND;
    float* aa_out   = xn_last + BD;
    float* bb_out   = aa_out + BD;
    float* pp_out   = bb_out + BD;
    float* xn2_last = pp_out + BD;

    // Workspace layout
    char* ws = (char*)d_ws;
    float* f_xn  = (float*)ws;                 ws += ND * 4;   // xn / xn2
    float* f_k   = (float*)ws;                 ws += ND * 4;
    float* f_v   = (float*)ws;                 ws += ND * 4;
    float* f_r   = (float*)ws;                 ws += ND * 4;   // r, later rc
    float* f_wkv = (float*)ws;                 ws += ND * 4;
    float* f_x1  = (float*)ws;                 ws += ND * 4;
    bf16*  h_a   = (bf16*)ws;                  ws += ND * 2;   // GEMM A panel
    bf16*  h_kc  = (bf16*)ws;                  ws += NF * 2;   // relu^2 result
    bf16*  h_w   = (bf16*)ws;                  /* ws += DF*2 */ // transposed weight

    const dim3 blk(256);
    const dim3 grid_rows(RN);                       // LN: one block per row
    const dim3 grid_nd((unsigned)(ND / 256));       // elementwise over N*D
    const dim3 grid_dd((unsigned)((RD * (size_t)RD + 255) / 256));
    const dim3 grid_df((unsigned)((DF + 255) / 256));
    const dim3 grid_g_d(RN / 16 / 8, RD / 64);      // GEMM Nn=1024
    const dim3 grid_g_f(RN / 16 / 8, RF / 64);      // GEMM Nn=4096
    const dim3 grid_bd((unsigned)(BD / 256));       // WKV channels

    // ---- ln1 ----
    ln_kernel<<<grid_rows, blk, 0, stream>>>(x, ln1_g, ln1_b, f_xn, xn_last);

    // ---- k = mix_k @ Wk ----
    convtrans_kernel<<<grid_dd, blk, 0, stream>>>(Wk, h_w, RD, RD);
    mix_kernel<<<grid_nd, blk, 0, stream>>>(f_xn, att_x, tm_mix_k, h_a);
    gemm_wmma_kernel<0><<<grid_g_d, blk, 0, stream>>>(h_a, h_w, RD, RD, f_k, nullptr, nullptr, nullptr);

    // ---- v = mix_v @ Wv ----
    convtrans_kernel<<<grid_dd, blk, 0, stream>>>(Wv, h_w, RD, RD);
    mix_kernel<<<grid_nd, blk, 0, stream>>>(f_xn, att_x, tm_mix_v, h_a);
    gemm_wmma_kernel<0><<<grid_g_d, blk, 0, stream>>>(h_a, h_w, RD, RD, f_v, nullptr, nullptr, nullptr);

    // ---- r = sigmoid(mix_r @ Wr) ----
    convtrans_kernel<<<grid_dd, blk, 0, stream>>>(Wr, h_w, RD, RD);
    mix_kernel<<<grid_nd, blk, 0, stream>>>(f_xn, att_x, tm_mix_r, h_a);
    gemm_wmma_kernel<1><<<grid_g_d, blk, 0, stream>>>(h_a, h_w, RD, RD, f_r, nullptr, nullptr, nullptr);

    // ---- WKV scan ----
    wkv_kernel<<<grid_bd, blk, 0, stream>>>(f_k, f_v, att_a, att_b, att_p,
                                            tm_decay, tm_first, f_wkv,
                                            aa_out, bb_out, pp_out);

    // ---- x1 = x + (r*wkv) @ Wo ----
    mul_bf16_kernel<<<grid_nd, blk, 0, stream>>>(f_r, f_wkv, h_a);
    convtrans_kernel<<<grid_dd, blk, 0, stream>>>(Wo, h_w, RD, RD);
    gemm_wmma_kernel<2><<<grid_g_d, blk, 0, stream>>>(h_a, h_w, RD, RD, f_x1, nullptr, x, nullptr);

    // ---- ln2 ----
    ln_kernel<<<grid_rows, blk, 0, stream>>>(f_x1, ln2_g, ln2_b, f_xn, xn2_last);

    // ---- kc = relu(mix_ck @ Ck)^2 -> bf16 ----
    convtrans_kernel<<<grid_df, blk, 0, stream>>>(Ck, h_w, RD, RF);
    mix_kernel<<<grid_nd, blk, 0, stream>>>(f_xn, ffn_x, cm_mix_k, h_a);
    gemm_wmma_kernel<3><<<grid_g_f, blk, 0, stream>>>(h_a, h_w, RD, RF, nullptr, h_kc, nullptr, nullptr);

    // ---- rc = sigmoid(mix_cr @ Cr) ----
    convtrans_kernel<<<grid_dd, blk, 0, stream>>>(Cr, h_w, RD, RD);
    mix_kernel<<<grid_nd, blk, 0, stream>>>(f_xn, ffn_x, cm_mix_r, h_a);
    gemm_wmma_kernel<1><<<grid_g_d, blk, 0, stream>>>(h_a, h_w, RD, RD, f_r, nullptr, nullptr, nullptr);

    // ---- out = x1 + rc * (kc @ Cv) ----
    convtrans_kernel<<<grid_df, blk, 0, stream>>>(Cv, h_w, RF, RD);
    gemm_wmma_kernel<4><<<grid_g_d, blk, 0, stream>>>(h_kc, h_w, RF, RD, out, nullptr, f_x1, f_r);
}
